// LocalAttention_31044023616477
// MI455X (gfx1250) — matrix operand order repaired and checked by reading, not compiled, not run on hardware
//
#include <hip/hip_runtime.h>
#include <math.h>

// f16 WMMA path: v_wmma_f32_16x16x32_f16, fp32 accumulate.
typedef __attribute__((ext_vector_type(16))) _Float16 v16h;
typedef __attribute__((ext_vector_type(8)))  _Float16 v8h;
typedef __attribute__((ext_vector_type(8)))  float    v8f;

#define WINDOW  128
#define DHEAD   64
#define JLEN    256      // window + look_backward*window
#define NWIN    32       // 4096 / 128
#define BATCH   128
#define QPITCH  72       // halfs; 144B rows (16B multiple)
#define KPITCH  72
#define VPITCH  272      // halfs; 544B rows
#define PPITCH  272

__device__ __forceinline__ float inv_freq(int c) {
    // 10000^(-c/32) = exp2(-c * log2(10000)/32)
    return exp2f((float)c * (-13.287712379549449f / 32.0f));
}

__global__ __launch_bounds__(256, 1)
void local_attn_f16wmma(const float* __restrict__ q,
                        const float* __restrict__ k,
                        const float* __restrict__ v,
                        float* __restrict__ out) {
    __shared__ __align__(16) _Float16 qs[WINDOW * QPITCH];   // 18.0 KB
    __shared__ __align__(16) _Float16 ks[JLEN   * KPITCH];   // 36.0 KB
    __shared__ __align__(16) _Float16 vt[DHEAD  * VPITCH];   // 34.0 KB (V transposed)
    __shared__ __align__(16) _Float16 ps[8 * 16 * PPITCH];   // 68.0 KB (per-wave P)

    const int w   = blockIdx.x;   // window index
    const int b   = blockIdx.y;   // batch index
    const int tid = threadIdx.x;

    const size_t batch_off = (size_t)b * (NWIN * WINDOW) * DHEAD;
    const float* qg = q + batch_off + (size_t)w * WINDOW * DHEAD;
    const float* kg = k + batch_off;   // key row j -> global pos (w-1)*128 + j
    const float* vg = v + batch_off;

    // ---- Phase 1: Q -> LDS with RoPE (t = 128 + row) and scale D^-0.5 ----
    for (int it = 0; it < 16; ++it) {
        int idx = tid + it * 256;            // 4096 (row, c) pairs, c in [0,32)
        int row = idx >> 5;
        int c   = idx & 31;
        float x0 = qg[row * DHEAD + c];
        float x1 = qg[row * DHEAD + c + 32];
        float f = (float)(WINDOW + row) * inv_freq(c);
        float sn, cs;
        __sincosf(f, &sn, &cs);
        const float scale = 0.125f;          // 64^-0.5
        qs[row * QPITCH + c]      = (_Float16)((x0 * cs - x1 * sn) * scale);
        qs[row * QPITCH + c + 32] = (_Float16)((x1 * cs + x0 * sn) * scale);
    }
    // ---- Phase 2: K -> LDS with RoPE (t = row), pad rows = -1 ----
    for (int it = 0; it < 32; ++it) {
        int idx = tid + it * 256;            // 8192 pairs
        int row = idx >> 5;
        int c   = idx & 31;
        int g = (w - 1) * WINDOW + row;
        float x0 = -1.0f, x1 = -1.0f;
        if (g >= 0) { x0 = kg[g * DHEAD + c]; x1 = kg[g * DHEAD + c + 32]; }
        float f = (float)row * inv_freq(c);
        float sn, cs;
        __sincosf(f, &sn, &cs);
        ks[row * KPITCH + c]      = (_Float16)(x0 * cs - x1 * sn);
        ks[row * KPITCH + c + 32] = (_Float16)(x1 * cs + x0 * sn);
    }
    // ---- Phase 3: V -> LDS transposed (vt[d][j]), pad rows = -1 ----
    for (int it = 0; it < 64; ++it) {
        int idx = tid + it * 256;            // 16384 elements
        int row = idx >> 6;                  // j
        int d   = idx & 63;
        int g = (w - 1) * WINDOW + row;
        float x = (g < 0) ? -1.0f : vg[g * DHEAD + d];
        vt[d * VPITCH + row] = (_Float16)x;
    }
    __syncthreads();

    const int wave = tid >> 5;
    const int lane = tid & 31;
    const int l16  = lane & 15;
    const int hi   = lane >> 4;
    const int row0 = wave * 16;              // this wave's Q rows

    // ---- Phase 4: S = Q·K^T  (16x256 per wave) ----
    // A-fragment layout (16x32 f16): lane row = lane%16; K octets {hi*8..+7, 16+hi*8..+7}
    v16h afrag[2];
    #pragma unroll
    for (int kc = 0; kc < 2; ++kc) {
        const _Float16* base = &qs[(row0 + l16) * QPITCH + kc * 32 + hi * 8];
        v8h lo = *(const v8h*)base;
        v8h hh = *(const v8h*)(base + 16);
        #pragma unroll
        for (int i = 0; i < 8; ++i) { afrag[kc][i] = lo[i]; afrag[kc][i + 8] = hh[i]; }
    }

    v8f acc[16] = {};
    #pragma unroll
    for (int jt = 0; jt < 16; ++jt) {
        // B-fragment (32x16): lane col = K-row j = jt*16 + lane%16; K octets {hi*8..+7, 16+hi*8..+7}
        const _Float16* krow = &ks[(jt * 16 + l16) * KPITCH];
        #pragma unroll
        for (int kc = 0; kc < 2; ++kc) {
            const _Float16* kb = krow + kc * 32 + hi * 8;
            v8h blo = *(const v8h*)kb;
            v8h bhi = *(const v8h*)(kb + 16);
            v16h bfrag;
            #pragma unroll
            for (int i = 0; i < 8; ++i) { bfrag[i] = blo[i]; bfrag[i + 8] = bhi[i]; }
            acc[jt] = __builtin_amdgcn_wmma_f32_16x16x32_f16(
                false, afrag[kc], false, bfrag, (short)0, acc[jt], false, false);
        }
    }

    // ---- Phase 5: mask + softmax (row-wise over 256 keys) ----
    // C/D layout: lane holds column n = lane%16; VGPR e holds row (e + 8*hi)
    const float NEG_MAX = -3.402823466e38f;
    float rmax[8], rsum[8];
    #pragma unroll
    for (int e = 0; e < 8; ++e) rmax[e] = NEG_MAX;

    #pragma unroll
    for (int jt = 0; jt < 16; ++jt) {
        int j = jt * 16 + l16;
        bool jvalid = (w > 0) || (j >= WINDOW);     // pad mask
        #pragma unroll
        for (int e = 0; e < 8; ++e) {
            int i = row0 + hi * 8 + e;              // query row in window
            bool ok = jvalid && (j <= WINDOW + i);  // keep t_q >= t_k
            float s = ok ? acc[jt][e] : NEG_MAX;
            acc[jt][e] = s;
            rmax[e] = fmaxf(rmax[e], s);
        }
    }
    #pragma unroll
    for (int e = 0; e < 8; ++e) {
        float m = rmax[e];
        #pragma unroll
        for (int off = 8; off >= 1; off >>= 1)      // reduce across 16 cols (stay in half)
            m = fmaxf(m, __shfl_xor(m, off, 32));
        rmax[e] = m;
        rsum[e] = 0.0f;
    }
    #pragma unroll
    for (int jt = 0; jt < 16; ++jt) {
        #pragma unroll
        for (int e = 0; e < 8; ++e) {
            float p = __expf(acc[jt][e] - rmax[e]); // masked -> exp(-huge) = 0
            acc[jt][e] = p;
            rsum[e] += p;
        }
    }
    #pragma unroll
    for (int e = 0; e < 8; ++e) {
        float s = rsum[e];
        #pragma unroll
        for (int off = 8; off >= 1; off >>= 1)
            s += __shfl_xor(s, off, 32);
        rsum[e] = 1.0f / s;
    }
    // P (f16) to per-wave LDS: converts D-layout -> A-fragment layout via memory
    #pragma unroll
    for (int jt = 0; jt < 16; ++jt) {
        #pragma unroll
        for (int e = 0; e < 8; ++e) {
            int r = hi * 8 + e;
            ps[(wave * 16 + r) * PPITCH + jt * 16 + l16] =
                (_Float16)(acc[jt][e] * rsum[e]);
        }
    }

    // ---- Phase 6: O = P·V  (16x64 per wave; K = 256 over 8 chunks of 32) ----
    v8f oacc[4] = {};
    #pragma unroll
    for (int jc = 0; jc < 8; ++jc) {
        v16h pa;
        {
            const _Float16* base = &ps[(wave * 16 + l16) * PPITCH + jc * 32 + hi * 8];
            v8h lo = *(const v8h*)base;
            v8h hh = *(const v8h*)(base + 16);
            #pragma unroll
            for (int i = 0; i < 8; ++i) { pa[i] = lo[i]; pa[i + 8] = hh[i]; }
        }
        #pragma unroll
        for (int dt = 0; dt < 4; ++dt) {
            const _Float16* vb = &vt[(dt * 16 + l16) * VPITCH + jc * 32 + hi * 8];
            v8h blo = *(const v8h*)vb;
            v8h bhi = *(const v8h*)(vb + 16);
            v16h bfrag;
            #pragma unroll
            for (int i = 0; i < 8; ++i) { bfrag[i] = blo[i]; bfrag[i + 8] = bhi[i]; }
            oacc[dt] = __builtin_amdgcn_wmma_f32_16x16x32_f16(
                false, pa, false, bfrag, (short)0, oacc[dt], false, false);
        }
    }

    // ---- Phase 7: store O (f32, coalesced: lanes 0..15 -> consecutive d) ----
    float* ob = out + batch_off + (size_t)(w * WINDOW + row0) * DHEAD;
    #pragma unroll
    for (int dt = 0; dt < 4; ++dt) {
        #pragma unroll
        for (int e = 0; e < 8; ++e) {
            ob[(hi * 8 + e) * DHEAD + dt * 16 + l16] = oacc[dt][e];
        }
    }
}

extern "C" void kernel_launch(void* const* d_in, const int* in_sizes, int n_in,
                              void* d_out, int out_size, void* d_ws, size_t ws_size,
                              hipStream_t stream) {
    const float* q = (const float*)d_in[0];
    const float* k = (const float*)d_in[1];
    const float* v = (const float*)d_in[2];
    float* out = (float*)d_out;
    dim3 grid(NWIN, BATCH);   // (window, batch) = (32, 128) -> 4096 workgroups
    local_attn_f16wmma<<<grid, dim3(256), 0, stream>>>(q, k, v, out);
    (void)in_sizes; (void)n_in; (void)out_size; (void)d_ws; (void)ws_size;
}
